// LLaVACLIP3DMultiModalProjector_77369540870729
// MI455X (gfx1250) — matrix operand order repaired and checked by reading, not compiled, not run on hardware
//
#include <hip/hip_runtime.h>

#define D_VIS  1024
#define D_CLIP 768
#define D_TXT  4096
#define BATCH  2
#define TFRM   16
#define FW     24
#define FH     24
#define LTOK   (FW * FH)            // 576
#define ROWS   (BATCH * TFRM * LTOK) // 18432
#define TOK_PER_B (TFRM * LTOK)      // 9216
#define OT 16
#define OW 12
#define OH 12
#define NOFF 18
#define POOL_TEMP 0.01f
#define NEGV (-1e4f)

typedef __attribute__((ext_vector_type(16))) __bf16 v16bf;
typedef __attribute__((ext_vector_type(8)))  float  v8f;
typedef int i32x4 __attribute__((vector_size(16)));

struct __align__(16) U4 { unsigned int a, b, c, d; };

// --- CDNA5 async global->LDS copy (ASYNCcnt) with sync fallback -------------
#if defined(__has_builtin) && __has_builtin(__builtin_amdgcn_global_load_async_to_lds_b128)
#define HAS_ASYNC_LDS 1
#else
#define HAS_ASYNC_LDS 0
#endif
#if defined(__has_builtin) && __has_builtin(__builtin_amdgcn_s_wait_asynccnt)
#define HAS_WAIT_ASYNC 1
#else
#define HAS_WAIT_ASYNC 0
#endif

#if HAS_ASYNC_LDS
#if HAS_WAIT_ASYNC
#define WAIT_ASYNC(N) __builtin_amdgcn_s_wait_asynccnt(N)
#else
#define WAIT_ASYNC(N) asm volatile("s_wait_asynccnt " #N ::: "memory")
#endif
#else
#define WAIT_ASYNC(N) do { } while (0)
#endif

__device__ __forceinline__ void cp16_g2l(const char* g, char* l) {
#if HAS_ASYNC_LDS
  __builtin_amdgcn_global_load_async_to_lds_b128(
      (__attribute__((address_space(1))) i32x4*)(unsigned long long)(const void*)g,
      (__attribute__((address_space(3))) i32x4*)(unsigned int)(unsigned long long)(void*)l,
      0, 0);
#else
  *(U4*)l = *(const U4*)g;
#endif
}

__device__ __forceinline__ unsigned short f2bf(float f) {
  unsigned int u = __float_as_uint(f);
  u += 0x7FFFu + ((u >> 16) & 1u);   // round-to-nearest-even
  return (unsigned short)(u >> 16);
}

// ---------------------------------------------------------------------------
// LayerNorm over D_VIS + dual bf16 cast: xbf = bf16(x), xnbf = bf16(LN(x))
// ---------------------------------------------------------------------------
__global__ void ln_cast_kernel(const float* __restrict__ x,
                               const float* __restrict__ gamma,
                               const float* __restrict__ beta,
                               unsigned short* __restrict__ xbf,
                               unsigned short* __restrict__ xnbf) {
  __shared__ float s1[256];
  __shared__ float s2[256];
  const int row = blockIdx.x;
  const int tid = threadIdx.x;
  const float* xr = x + (size_t)row * D_VIS;
  float v[4];
  float sum = 0.f, sq = 0.f;
#pragma unroll
  for (int i = 0; i < 4; ++i) {
    v[i] = xr[tid + i * 256];
    sum += v[i];
    sq  += v[i] * v[i];
  }
  s1[tid] = sum; s2[tid] = sq;
  __syncthreads();
  for (int st = 128; st > 0; st >>= 1) {
    if (tid < st) { s1[tid] += s1[tid + st]; s2[tid] += s2[tid + st]; }
    __syncthreads();
  }
  const float mean = s1[0] * (1.0f / D_VIS);
  const float var  = s2[0] * (1.0f / D_VIS) - mean * mean;
  const float rstd = rsqrtf(var + 1e-5f);
#pragma unroll
  for (int i = 0; i < 4; ++i) {
    const int j = tid + i * 256;
    const float xn = (v[i] - mean) * rstd * gamma[j] + beta[j];
    xbf [(size_t)row * D_VIS + j] = f2bf(v[i]);
    xnbf[(size_t)row * D_VIS + j] = f2bf(xn);
  }
}

// ---------------------------------------------------------------------------
// Weight transpose + bf16 cast: wT[n*K+k] = bf16(w[k*N+n])  (w is [K,N])
// ---------------------------------------------------------------------------
__global__ void transpose_bf16_kernel(const float* __restrict__ w,
                                      unsigned short* __restrict__ wT,
                                      int K, int N) {
  const size_t total = (size_t)K * N;
  for (size_t idx = (size_t)blockIdx.x * blockDim.x + threadIdx.x; idx < total;
       idx += (size_t)gridDim.x * blockDim.x) {
    const size_t n = idx / (size_t)K;
    const size_t k = idx % (size_t)K;
    wT[idx] = f2bf(w[k * (size_t)N + n]);
  }
}

// ---------------------------------------------------------------------------
// txt = normalize(text_pooled @ w_text_proj)  — tiny, one block per batch row
// ---------------------------------------------------------------------------
__global__ void txt_kernel(const float* __restrict__ tp,
                           const float* __restrict__ wt,
                           float* __restrict__ txtn) {
  __shared__ float so[D_CLIP];
  __shared__ float sq[256];
  const int b = blockIdx.x, tid = threadIdx.x;
  const float* t = tp + (size_t)b * D_CLIP;
  float q = 0.f;
  for (int c = tid; c < D_CLIP; c += 256) {
    float acc = 0.f;
    for (int j = 0; j < D_CLIP; ++j) acc = fmaf(t[j], wt[(size_t)j * D_CLIP + c], acc);
    so[c] = acc;
    q += acc * acc;
  }
  sq[tid] = q;
  __syncthreads();
  for (int st = 128; st > 0; st >>= 1) {
    if (tid < st) sq[tid] += sq[tid + st];
    __syncthreads();
  }
  const float rn = rsqrtf(sq[0]);
  for (int c = tid; c < D_CLIP; c += 256) txtn[(size_t)b * D_CLIP + c] = so[c] * rn;
}

// ---------------------------------------------------------------------------
// Per-token cosine logit: logits[g] = dot(img_g, txt_b)/||img_g|| * exp(ls)
// ---------------------------------------------------------------------------
__global__ void logits_kernel(const float* __restrict__ img,
                              const float* __restrict__ txtn,
                              const float* __restrict__ lscale,
                              float* __restrict__ logits) {
  __shared__ float sd[256];
  __shared__ float sq[256];
  const int g = blockIdx.x, tid = threadIdx.x;
  const int b = g / TOK_PER_B;
  const float* r = img + (size_t)g * D_CLIP;
  const float* t = txtn + (size_t)b * D_CLIP;
  float d = 0.f, q = 0.f;
  for (int j = tid; j < D_CLIP; j += 256) {
    const float v = r[j];
    d = fmaf(v, t[j], d);
    q = fmaf(v, v, q);
  }
  sd[tid] = d; sq[tid] = q;
  __syncthreads();
  for (int st = 128; st > 0; st >>= 1) {
    if (tid < st) { sd[tid] += sd[tid + st]; sq[tid] += sq[tid + st]; }
    __syncthreads();
  }
  if (tid == 0) logits[g] = sd[0] * rsqrtf(sq[0]) * expf(lscale[0]);
}

// ---------------------------------------------------------------------------
// Softmax over the 9216 tokens of one batch element -> gating weights tv
// ---------------------------------------------------------------------------
__global__ void softmax_kernel(const float* __restrict__ logits,
                               float* __restrict__ tv) {
  __shared__ float s[1024];
  const int b = blockIdx.x, tid = threadIdx.x;
  const float* l = logits + (size_t)b * TOK_PER_B;
  float m = -3.4e38f;
  for (int j = tid; j < TOK_PER_B; j += 1024) m = fmaxf(m, l[j]);
  s[tid] = m;
  __syncthreads();
  for (int st = 512; st > 0; st >>= 1) {
    if (tid < st) s[tid] = fmaxf(s[tid], s[tid + st]);
    __syncthreads();
  }
  m = s[0];
  __syncthreads();
  float sum = 0.f;
  for (int j = tid; j < TOK_PER_B; j += 1024) sum += expf(l[j] - m);
  s[tid] = sum;
  __syncthreads();
  for (int st = 512; st > 0; st >>= 1) {
    if (tid < st) s[tid] += s[tid + st];
    __syncthreads();
  }
  const float inv = 1.f / s[0];
  for (int j = tid; j < TOK_PER_B; j += 1024)
    tv[(size_t)b * TOK_PER_B + j] = expf(l[j] - m) * inv;
}

// ---------------------------------------------------------------------------
// bf16 WMMA GEMM with double-buffered async LDS staging:
//   C[M,N] = act(A[M,K] @ BT[N,K]^T + bias),  K/N compile-time.
// Block = 256 threads = 8 waves (4 M x 2 N), block tile 128x128, K step 32.
// Per K-slab each thread issues 4 GLOBAL_LOAD_ASYNC_TO_LDS_B128 (ASYNCcnt);
// slabs stored row-major (64B per row of 32 K) so WMMA fragments read as
// contiguous ds_load_b128:
//   A frag: row (tileM+l15), bytes hl*16 and hl*16+32  (K {hl*8..}, {16+hl*8..})
//   B frag: row (tileN+l15), bytes hl*16 and hl*16+32  (K {hl*8..}, {16+hl*8..})
// ---------------------------------------------------------------------------
template <int K, int N, int OUT_BF16, int GELU>
__global__ void __launch_bounds__(256, 1)
gemm_wmma_kernel(const unsigned short* __restrict__ A,
                 const unsigned short* __restrict__ BT,
                 const float* __restrict__ bias,
                 float* __restrict__ Cf,
                 unsigned short* __restrict__ Cb) {
  __shared__ __align__(16) unsigned short As[2][128 * 32];
  __shared__ __align__(16) unsigned short Bs[2][128 * 32];

  const int tid  = threadIdx.x;
  const int lane = tid & 31;
  const int wave = tid >> 5;
  const int l15  = lane & 15;
  const int hl   = lane >> 4;
  const int blockM = blockIdx.y * 128;
  const int blockN = blockIdx.x * 128;

  // --- copy-lane assignment: 512 16B chunks per slab, 2 per thread ---------
  int crow[2], cq[2];
  const char* gA[2];
  const char* gB[2];
#pragma unroll
  for (int t = 0; t < 2; ++t) {
    const int c = tid + t * 256;
    crow[t] = c >> 2;       // 0..127
    cq[t]   = c & 3;        // 16B quad within 64B row
    gA[t] = (const char*)A  + (size_t)(blockM + crow[t]) * K * 2 + cq[t] * 16;
    gB[t] = (const char*)BT + (size_t)(blockN + crow[t]) * K * 2 + cq[t] * 16;
  }
  auto issue_slab = [&](int buf, int k0c) {
    char* sA = (char*)&As[buf][0];
    char* sB = (char*)&Bs[buf][0];
#pragma unroll
    for (int t = 0; t < 2; ++t) {
      const int loff = crow[t] * 64 + cq[t] * 16;
      cp16_g2l(gA[t] + (size_t)k0c * 2, sA + loff);
      cp16_g2l(gB[t] + (size_t)k0c * 2, sB + loff);
    }
  };

  v8f acc[2][4] = {};
  union FragU { U4 q[2]; v16bf v; };

  issue_slab(0, 0);

#pragma unroll 1
  for (int k0 = 0; k0 < K; k0 += 32) {
    const int cur = (k0 >> 5) & 1;
    const bool has_next = (k0 + 32) < K;
    if (has_next) {
      issue_slab(cur ^ 1, k0 + 32);
      WAIT_ASYNC(4);          // current slab's 4 copies done; next 4 in flight
    } else {
      WAIT_ASYNC(0);
    }
    __syncthreads();          // all waves' slab data visible in LDS

    const char* sA = (const char*)&As[cur][0];
    const char* sB = (const char*)&Bs[cur][0];
    FragU af[2], bfr[4];
#pragma unroll
    for (int i = 0; i < 2; ++i) {
      const int row = (wave & 3) * 32 + i * 16 + l15;
      af[i].q[0] = *(const U4*)(sA + row * 64 + hl * 16);
      af[i].q[1] = *(const U4*)(sA + row * 64 + hl * 16 + 32);
    }
#pragma unroll
    for (int j = 0; j < 4; ++j) {
      const int row = (wave >> 2) * 64 + j * 16 + l15;
      bfr[j].q[0] = *(const U4*)(sB + row * 64 + hl * 16);
      bfr[j].q[1] = *(const U4*)(sB + row * 64 + hl * 16 + 32);
    }
#pragma unroll
    for (int i = 0; i < 2; ++i)
#pragma unroll
      for (int j = 0; j < 4; ++j)
        acc[i][j] = __builtin_amdgcn_wmma_f32_16x16x32_bf16(
            false, af[i].v, false, bfr[j].v, (short)0, acc[i][j], false, false);

    __syncthreads();          // done reading buf[cur] before it is refilled
  }

  // Epilogue: C/D layout — VGPR r: lanes 0-15 -> M=r, lanes 16-31 -> M=r+8
  const int waveM = blockM + (wave & 3) * 32;
  const int waveN = blockN + (wave >> 2) * 64;
#pragma unroll
  for (int i = 0; i < 2; ++i) {
#pragma unroll
    for (int j = 0; j < 4; ++j) {
      const int n = waveN + j * 16 + l15;
      const float bv = bias ? bias[n] : 0.f;
#pragma unroll
      for (int r = 0; r < 8; ++r) {
        const int m = waveM + i * 16 + r + hl * 8;
        float v = acc[i][j][r] + bv;
        if (GELU) v = 0.5f * v * (1.f + erff(v * 0.70710678118654752f));
        if (OUT_BF16) Cb[(size_t)m * N + n] = f2bf(v);
        else          Cf[(size_t)m * N + n] = v;
      }
    }
  }
}

// ---------------------------------------------------------------------------
// Softmax-weighted 3D pool: one block per output cell (b, ot, ow, oh).
// 18 kernel offsets, softmax(tv/temp) over the window, weighted sum over D.
// ---------------------------------------------------------------------------
__global__ void pool_kernel(const float* __restrict__ h2,
                            const float* __restrict__ tv,
                            float* __restrict__ out) {
  __shared__ float swt[NOFF];
  __shared__ int   stok[NOFF];
  const int idx = blockIdx.x;
  const int oh = idx % OH;
  const int ow = (idx / OH) % OW;
  const int ot = (idx / (OH * OW)) % OT;
  const int b  = idx / (OH * OW * OT);
  const int tid = threadIdx.x;

  if (tid < NOFF) {
    const int dt = tid / 9, rem = tid % 9, dw = rem / 3, dh = rem % 3;
    const int t = ot + dt;           // stride 1, end-pad 1
    const int w = ow * 2 + dw;       // stride 2, end-pad 1
    const int h = oh * 2 + dh;       // stride 2, end-pad 1
    const bool inb = (t < TFRM) && (w < FW) && (h < FH);
    const int tok = ((b * TFRM + t) * FW + w) * FH + h;
    swt[tid]  = inb ? tv[tok] * (1.f / POOL_TEMP) : (NEGV / POOL_TEMP);
    stok[tid] = inb ? tok : -1;
  }
  __syncthreads();
  if (tid == 0) {
    float m = swt[0];
    for (int k = 1; k < NOFF; ++k) m = fmaxf(m, swt[k]);
    float s = 0.f;
    for (int k = 0; k < NOFF; ++k) { swt[k] = expf(swt[k] - m); s += swt[k]; }
    const float inv = 1.f / s;
    for (int k = 0; k < NOFF; ++k) swt[k] *= inv;
  }
  __syncthreads();

  const size_t obase = (size_t)idx * D_TXT;
  for (int d = tid; d < D_TXT; d += 256) {
    float acc = 0.f;
#pragma unroll
    for (int k = 0; k < NOFF; ++k) {
      const int tok = stok[k];
      if (tok >= 0) acc = fmaf(swt[k], h2[(size_t)tok * D_TXT + d], acc);
    }
    out[obase + d] = acc;
  }
}

// ---------------------------------------------------------------------------
extern "C" void kernel_launch(void* const* d_in, const int* in_sizes, int n_in,
                              void* d_out, int out_size, void* d_ws, size_t ws_size,
                              hipStream_t stream) {
  const float* x      = (const float*)d_in[0];   // [32,576,1024]
  const float* tp     = (const float*)d_in[1];   // [2,768]
  const float* gamma  = (const float*)d_in[2];   // [1024]
  const float* beta   = (const float*)d_in[3];   // [1024]
  const float* wvis   = (const float*)d_in[4];   // [1024,768]
  const float* wtxt   = (const float*)d_in[5];   // [768,768]
  const float* lscale = (const float*)d_in[6];   // [1]
  const float* w1     = (const float*)d_in[7];   // [1024,4096]
  const float* b1     = (const float*)d_in[8];   // [4096]
  const float* w2     = (const float*)d_in[9];   // [4096,4096]
  const float* b2     = (const float*)d_in[10];  // [4096]
  float* out = (float*)d_out;                    // [2,2304,4096]
  (void)in_sizes; (void)n_in; (void)out_size; (void)ws_size;

  char* ws = (char*)d_ws;
  size_t off = 0;
  auto alloc = [&](size_t bytes) -> void* {
    off = (off + 255) & ~(size_t)255;
    void* p = ws + off;
    off += bytes;
    return p;
  };
  unsigned short* xbf   = (unsigned short*)alloc((size_t)ROWS * D_VIS * 2);
  unsigned short* xnbf  = (unsigned short*)alloc((size_t)ROWS * D_VIS * 2);
  unsigned short* wvisT = (unsigned short*)alloc((size_t)D_CLIP * D_VIS * 2);
  unsigned short* w1T   = (unsigned short*)alloc((size_t)D_TXT * D_VIS * 2);
  unsigned short* w2T   = (unsigned short*)alloc((size_t)D_TXT * D_TXT * 2);
  float*          img   = (float*)alloc((size_t)ROWS * D_CLIP * 4);
  float*          txtn  = (float*)alloc((size_t)BATCH * D_CLIP * 4);
  float*          logit = (float*)alloc((size_t)BATCH * TOK_PER_B * 4);
  float*          tv    = (float*)alloc((size_t)BATCH * TOK_PER_B * 4);
  unsigned short* h1bf  = (unsigned short*)alloc((size_t)ROWS * D_TXT * 2);
  float*          h2    = (float*)alloc((size_t)ROWS * D_TXT * 4);

  // Prep: LN + bf16 casts, bf16 weight transposes
  ln_cast_kernel<<<ROWS, 256, 0, stream>>>(x, gamma, beta, xbf, xnbf);
  transpose_bf16_kernel<<<(D_VIS * D_CLIP) / 256, 256, 0, stream>>>(wvis, wvisT, D_VIS, D_CLIP);
  transpose_bf16_kernel<<<(D_VIS * D_TXT) / 256, 256, 0, stream>>>(w1, w1T, D_VIS, D_TXT);
  transpose_bf16_kernel<<<(D_TXT * D_TXT) / 256, 256, 0, stream>>>(w2, w2T, D_TXT, D_TXT);

  // CLIP visual projection GEMM (f32 out)
  dim3 g1(D_CLIP / 128, ROWS / 128);
  gemm_wmma_kernel<D_VIS, D_CLIP, 0, 0><<<g1, 256, 0, stream>>>(xnbf, wvisT, nullptr, img, nullptr);
  // Text branch + gating softmax
  txt_kernel<<<BATCH, 256, 0, stream>>>(tp, wtxt, txtn);
  logits_kernel<<<ROWS, 256, 0, stream>>>(img, txtn, lscale, logit);
  softmax_kernel<<<BATCH, 1024, 0, stream>>>(logit, tv);

  // MM projector MLP: fc1 (gelu, bf16 out) then fc2 (f32 out)
  dim3 g2(D_TXT / 128, ROWS / 128);
  gemm_wmma_kernel<D_VIS, D_TXT, 1, 1><<<g2, 256, 0, stream>>>(xbf, w1T, b1, nullptr, h1bf);
  gemm_wmma_kernel<D_TXT, D_TXT, 0, 0><<<g2, 256, 0, stream>>>(h1bf, w2T, b2, h2, nullptr);

  // Softmax-weighted 3D pooling -> output
  pool_kernel<<<BATCH * OT * OW * OH, 256, 0, stream>>>(h2, tv, out);
}
